// GraphAttention_48833778155963
// MI455X (gfx1250) — matrix operand order repaired and checked by reading, not compiled, not run on hardware
//
#include <hip/hip_runtime.h>

// CDNA5 / gfx1250: wave32, v_wmma_f32_16x16x32_bf16 (f32 accum).
typedef __attribute__((ext_vector_type(16))) __bf16 v16bf;
typedef __attribute__((ext_vector_type(8)))  float  v8f;

#define B_  8
#define D_  384
#define NQ_ 2048
#define NK_ 2048
#define K_  8
#define NEG_SLOPE 0.2f
#define FS_ (D_ / 32)    // 12 f-steps of 32
#define NT_ (NQ_ / 16)   // 128 n-tiles of 16
#define DT_ (D_ / 16)    // 24 d-tiles of 16

// f32 -> bf16 round-to-nearest-even, two packed into one dword
__device__ __forceinline__ unsigned bf16rne(float f) {
    unsigned u = __builtin_bit_cast(unsigned, f);
    return (u + 0x7FFFu + ((u >> 16) & 1u)) >> 16;
}
__device__ __forceinline__ unsigned pack2(float lo, float hi) {
    return bf16rne(lo) | (bf16rne(hi) << 16);
}

union TileU { uint4 q[2]; v16bf v; };

// ---------------------------------------------------------------------------
// Pack X (B, D, N) f32 -> bf16 B-matrix tiles in WMMA per-lane layout.
// Unit = (b, n-tile t, f-step fs): 32 lanes x 8 dwords (pairs along k).
// Lane: n = t*16 + (lane&15); dword v holds k = 16*(lane>>4) + 2v, +1.
// Writes are per-lane contiguous 32B (2 x b128), matching GEMM consumption.
// ---------------------------------------------------------------------------
__global__ void pack_x_kernel(const float* __restrict__ X, unsigned* __restrict__ Xp) {
    const int lane = threadIdx.x & 31;
    const int wave = threadIdx.x >> 5;
    const int half = lane >> 4;
    const int ml   = lane & 15;
    const int t    = blockIdx.x * 8 + wave;
    const int fs   = blockIdx.y;
    const int b    = blockIdx.z;
    const int n    = t * 16 + ml;
    const int f0   = fs * 32;
    const float* Xb = X + (size_t)b * D_ * NQ_;

    unsigned w[8];
#pragma unroll
    for (int v = 0; v < 8; ++v) {
        const int k = 16 * half + 2 * v;
        w[v] = pack2(Xb[(size_t)(f0 + k) * NQ_ + n],
                     Xb[(size_t)(f0 + k + 1) * NQ_ + n]);
    }
    unsigned* o = Xp + (((size_t)(b * NT_ + t) * FS_ + fs) * 256) + lane * 8;
    *(uint4*)(o)     = make_uint4(w[0], w[1], w[2], w[3]);
    *(uint4*)(o + 4) = make_uint4(w[4], w[5], w[6], w[7]);
}

// ---------------------------------------------------------------------------
// Pack W (D, 2D) f32 -> bf16 A-matrix tiles, both modes:
//   mode 0: A = W[:, 0:D]          mode 1: A = W[:, D:2D] - W[:, 0:D]
// Lane: d = dt*16 + (lane&15); dword v holds k = 16*(lane>>4) + 2v, +1.
// ---------------------------------------------------------------------------
__global__ void pack_w_kernel(const float* __restrict__ W, unsigned* __restrict__ Ap) {
    const int lane = threadIdx.x & 31;
    const int wave = threadIdx.x >> 5;
    const int half = lane >> 4;
    const int ml   = lane & 15;
    const int dt   = blockIdx.x * 8 + wave;   // 0..23
    const int fs   = blockIdx.y;
    const int mode = blockIdx.z;
    const int d    = dt * 16 + ml;
    const int f0   = fs * 32;
    const float* wr = W + (size_t)d * (2 * D_) + f0;

    unsigned w[8];
#pragma unroll
    for (int v = 0; v < 8; ++v) {
        const int k = 16 * half + 2 * v;
        float a0, a1;
        if (mode == 0) {
            a0 = wr[k];
            a1 = wr[k + 1];
        } else {
            a0 = wr[D_ + k]     - wr[k];
            a1 = wr[D_ + k + 1] - wr[k + 1];
        }
        w[v] = pack2(a0, a1);
    }
    unsigned* o = Ap + (((size_t)(mode * DT_ + dt) * FS_ + fs) * 256) + lane * 8;
    *(uint4*)(o)     = make_uint4(w[0], w[1], w[2], w[3]);
    *(uint4*)(o + 4) = make_uint4(w[4], w[5], w[6], w[7]);
}

// ---------------------------------------------------------------------------
// GEMM: Out[b][n][d] = sum_f A[d][f] * X[b][f][n]  (+ bias if DO_BIAS)
// Pre-packed operands: hot loop = 10 x global_load_b128 + 4 x v_wmma per wave
// per f-step. Each wave: 1 n-tile x 4 d-tiles (B-tile amortized 4x).
// No divergence anywhere -> EXEC all-ones as WMMA requires.
// ---------------------------------------------------------------------------
template <bool DO_BIAS>
__global__ void gemm_kernel(const unsigned* __restrict__ Xp,
                            const unsigned* __restrict__ Ap,
                            const float* __restrict__ bias,
                            float* __restrict__ Out) {
    const int lane = threadIdx.x & 31;
    const int wave = threadIdx.x >> 5;
    const int half = lane >> 4;
    const int ml   = lane & 15;
    const int b    = blockIdx.z;
    const int t    = blockIdx.x * 8 + wave;   // n-tile
    const int dg   = blockIdx.y;              // d-group of 4 tiles (0..5)

    const uint4* xp = (const uint4*)(Xp + ((size_t)(b * NT_ + t) * FS_) * 256) + lane * 2;
    const uint4* ap = (const uint4*)(Ap + ((size_t)(dg * 4) * FS_) * 256) + lane * 2;

    v8f acc[4] = {};

    for (int fs = 0; fs < FS_; ++fs) {
        TileU bu;
        bu.q[0] = xp[0];
        bu.q[1] = xp[1];
        xp += 64;                              // 256 dwords per f-step
#pragma unroll
        for (int s = 0; s < 4; ++s) {
            TileU au;
            const uint4* aps = ap + (size_t)s * FS_ * 64;
            au.q[0] = aps[0];
            au.q[1] = aps[1];
            acc[s] = __builtin_amdgcn_wmma_f32_16x16x32_bf16(
                false, au.v, false, bu.v, (short)0, acc[s], false, false);
        }
        ap += 64;
    }

    // C/D layout: lane col n=ml, rows m = 8*half + r -> 8 contiguous d's
    float* ob = Out + (size_t)b * NQ_ * D_ + (size_t)(t * 16 + ml) * D_;
#pragma unroll
    for (int s = 0; s < 4; ++s) {
        const int dbase = dg * 64 + 16 * s + 8 * half;
        float4 lo = make_float4(acc[s][0], acc[s][1], acc[s][2], acc[s][3]);
        float4 hi = make_float4(acc[s][4], acc[s][5], acc[s][6], acc[s][7]);
        if (DO_BIAS) {
            const float4 b0 = *(const float4*)(bias + dbase);
            const float4 b1 = *(const float4*)(bias + dbase + 4);
            lo.x += b0.x; lo.y += b0.y; lo.z += b0.z; lo.w += b0.w;
            hi.x += b1.x; hi.y += b1.y; hi.z += b1.z; hi.w += b1.w;
        }
        *(float4*)(ob + dbase)     = lo;
        *(float4*)(ob + dbase + 4) = hi;
    }
}

// ---------------------------------------------------------------------------
// 3D KNN, top-8 smallest squared distance per query.
// ---------------------------------------------------------------------------
__global__ void knn_kernel(const float* __restrict__ qc,
                           const float* __restrict__ kc,
                           int* __restrict__ idx) {
    __shared__ float sx[256], sy[256], sz[256];
    const int b = blockIdx.y;
    const int q = blockIdx.x * 256 + threadIdx.x;

    const float* qcb = qc + (size_t)b * 3 * NQ_;
    const float qx = qcb[q];
    const float qy = qcb[NQ_ + q];
    const float qz = qcb[2 * NQ_ + q];

    float bd[K_];
    int   bi[K_];
#pragma unroll
    for (int i = 0; i < K_; ++i) { bd[i] = 3.4e38f; bi[i] = 0; }

    const float* kcb = kc + (size_t)b * 3 * NK_;
    for (int tbase = 0; tbase < NK_; tbase += 256) {
        __syncthreads();
        const int j = tbase + threadIdx.x;
        sx[threadIdx.x] = kcb[j];
        sy[threadIdx.x] = kcb[NK_ + j];
        sz[threadIdx.x] = kcb[2 * NK_ + j];
        __syncthreads();
        for (int u = 0; u < 256; ++u) {
            const float dx = qx - sx[u];
            const float dy = qy - sy[u];
            const float dz = qz - sz[u];
            const float d = dx * dx + dy * dy + dz * dz;
            if (d < bd[K_ - 1]) {
                int pos = K_ - 1;
#pragma unroll
                for (int i = K_ - 2; i >= 0; --i) {
                    if (d < bd[i]) { bd[i + 1] = bd[i]; bi[i + 1] = bi[i]; pos = i; }
                }
                bd[pos] = d; bi[pos] = tbase + u;
            }
        }
    }
    int* o = idx + ((size_t)b * NQ_ + q) * K_;
#pragma unroll
    for (int i = 0; i < K_; ++i) o[i] = bi[i];
}

// ---------------------------------------------------------------------------
// out[b][q][d] = lrelu( max_k ( Ykey[b][idx[b,q,k]][d] + C[b][q][d] ) )
// (leaky-ReLU is monotone increasing -> commutes with max.)
// ---------------------------------------------------------------------------
__global__ void finalize_kernel(const float* __restrict__ Ykey,
                                const float* __restrict__ Cc,
                                const int* __restrict__ idx,
                                float* __restrict__ out) {
    const int tid = blockIdx.x * blockDim.x + threadIdx.x;   // = bq*(D/4) + d4
    const int d4 = tid % (D_ / 4);
    const int bq = tid / (D_ / 4);
    const int b  = bq / NQ_;

    const int* ip = idx + (size_t)bq * K_;
    const float4 c = ((const float4*)(Cc + (size_t)bq * D_))[d4];
    const float* yb = Ykey + (size_t)b * NK_ * D_;

    float4 m = make_float4(-3.4e38f, -3.4e38f, -3.4e38f, -3.4e38f);
#pragma unroll
    for (int k = 0; k < K_; ++k) {
        const float4 y = ((const float4*)(yb + (size_t)ip[k] * D_))[d4];
        m.x = fmaxf(m.x, y.x + c.x);
        m.y = fmaxf(m.y, y.y + c.y);
        m.z = fmaxf(m.z, y.z + c.z);
        m.w = fmaxf(m.w, y.w + c.w);
    }
    m.x = (m.x >= 0.f) ? m.x : NEG_SLOPE * m.x;
    m.y = (m.y >= 0.f) ? m.y : NEG_SLOPE * m.y;
    m.z = (m.z >= 0.f) ? m.z : NEG_SLOPE * m.z;
    m.w = (m.w >= 0.f) ? m.w : NEG_SLOPE * m.w;
    ((float4*)out)[tid] = m;
}

// ---------------------------------------------------------------------------
extern "C" void kernel_launch(void* const* d_in, const int* in_sizes, int n_in,
                              void* d_out, int out_size, void* d_ws, size_t ws_size,
                              hipStream_t stream) {
    const float* qc   = (const float*)d_in[0];   // (B,3,NQ)
    const float* qf   = (const float*)d_in[1];   // (B,D,NQ)
    const float* kc   = (const float*)d_in[2];   // (B,3,NK)
    const float* kf   = (const float*)d_in[3];   // (B,D,NK)
    const float* W    = (const float*)d_in[4];   // (D,2D)
    const float* bias = (const float*)d_in[5];   // (D,)
    float* out = (float*)d_out;                  // (B,NQ,D)

    char* ws = (char*)d_ws;
    size_t off = 0;
    auto take = [&](size_t bytes) {
        void* p = ws + off;
        off = (off + bytes + 255) & ~(size_t)255;
        return p;
    };
    int*      idx  = (int*)take((size_t)B_ * NQ_ * K_ * sizeof(int));
    float*    Ykey = (float*)take((size_t)B_ * NK_ * D_ * sizeof(float));
    float*    Cc   = (float*)take((size_t)B_ * NQ_ * D_ * sizeof(float));
    unsigned* Ap   = (unsigned*)take((size_t)2 * DT_ * FS_ * 256 * sizeof(unsigned));
    unsigned* XpK  = (unsigned*)take((size_t)B_ * NT_ * FS_ * 256 * sizeof(unsigned));
    unsigned* XpQ  = (unsigned*)take((size_t)B_ * NT_ * FS_ * 256 * sizeof(unsigned));

    pack_w_kernel<<<dim3(DT_ / 8, FS_, 2), 256, 0, stream>>>(W, Ap);
    pack_x_kernel<<<dim3(NT_ / 8, FS_, B_), 256, 0, stream>>>(kf, XpK);
    pack_x_kernel<<<dim3(NT_ / 8, FS_, B_), 256, 0, stream>>>(qf, XpQ);
    knn_kernel<<<dim3(NQ_ / 256, B_), 256, 0, stream>>>(qc, kc, idx);

    const unsigned* Ap0 = Ap;                                  // mode 0: W1
    const unsigned* Ap1 = Ap + (size_t)DT_ * FS_ * 256;        // mode 1: W2-W1
    gemm_kernel<false><<<dim3(NT_ / 8, DT_ / 4, B_), 256, 0, stream>>>(XpK, Ap0, nullptr, Ykey);
    gemm_kernel<true ><<<dim3(NT_ / 8, DT_ / 4, B_), 256, 0, stream>>>(XpQ, Ap1, bias, Cc);

    finalize_kernel<<<(B_ * NQ_ * (D_ / 4)) / 256, 256, 0, stream>>>(Ykey, Cc, idx, out);
}
